// FuncMod_40484361732580
// MI455X (gfx1250) — matrix operand order repaired and checked by reading, not compiled, not run on hardware
//
#include <hip/hip_runtime.h>
#include <hip/hip_bf16.h>

// Sizes from the reference
#define B_SZ  16384
#define IN_SZ 512
#define CH_SZ 1024
#define ED_SZ 4096
#define NC_SZ 8
#define NE_SZ 512
#define D_SZ  512

typedef __bf16 bf16_t;
typedef __attribute__((ext_vector_type(8)))  __bf16 v8bf;
typedef __attribute__((ext_vector_type(16))) __bf16 v16bf;
typedef __attribute__((ext_vector_type(8)))  float  v8f;

static __device__ __forceinline__ v16bf frag16(const bf16_t* p0, const bf16_t* p1) {
  v8bf lo = *(const v8bf*)p0;
  v8bf hi = *(const v8bf*)p1;
  return __builtin_shufflevector(lo, hi, 0,1,2,3,4,5,6,7,8,9,10,11,12,13,14,15);
}

// 16-byte async DMA: global -> LDS, tracked by ASYNCcnt (CDNA5 path)
static __device__ __forceinline__ void async_ld16(unsigned int lds_off, const void* gptr) {
  asm volatile("global_load_async_to_lds_b128 %0, %1, off"
               :: "v"(lds_off), "v"((unsigned long long)(uintptr_t)gptr)
               : "memory");
}

// ---------------------------------------------------------------- converts
__global__ __launch_bounds__(256) void k_f32_to_bf16(const float* __restrict__ src,
                                                     bf16_t* __restrict__ dst, int n) {
  for (int i = blockIdx.x * blockDim.x + threadIdx.x; i < n; i += gridDim.x * blockDim.x)
    dst[i] = (bf16_t)src[i];
}

// W_x2 [63,1024] -> padded [64,1024] bf16 (row 63 = 0)
__global__ __launch_bounds__(256) void k_pad_wx2(const float* __restrict__ src,
                                                 bf16_t* __restrict__ dst) {
  int i = blockIdx.x * blockDim.x + threadIdx.x;     // over 64*1024
  if (i >= 64 * CH_SZ) return;
  int r = i >> 10, c = i & (CH_SZ - 1);
  dst[i] = (bf16_t)((r < 63) ? src[r * CH_SZ + c] : 0.0f);
}

// embed [NC, D, NE] -> eT [NC, NE, D] (bf16 + f32)
__global__ __launch_bounds__(256) void k_embed_T(const float* __restrict__ embed,
                                                 bf16_t* __restrict__ eT_bf,
                                                 float* __restrict__ eT_f) {
  int i = blockIdx.x * blockDim.x + threadIdx.x;     // over NC*D*NE
  if (i >= NC_SZ * D_SZ * NE_SZ) return;
  int c = i / (D_SZ * NE_SZ);
  int rem = i - c * D_SZ * NE_SZ;
  int d = rem / NE_SZ;
  int n = rem - d * NE_SZ;
  float v = embed[i];
  size_t o = (size_t)c * NE_SZ * D_SZ + (size_t)n * D_SZ + d;
  eT_bf[o] = (bf16_t)v;
  eT_f[o] = v;
}

// e_sq[c][n] = sum_d eT_f[c][n][d]^2
__global__ __launch_bounds__(256) void k_esq(const float* __restrict__ eT_f,
                                             float* __restrict__ e_sq) {
  int i = blockIdx.x * blockDim.x + threadIdx.x;     // over NC*NE
  if (i >= NC_SZ * NE_SZ) return;
  const float* pp = eT_f + (size_t)i * D_SZ;
  float s = 0.f;
  for (int d = 0; d < D_SZ; ++d) s += pp[d] * pp[d];
  e_sq[i] = s;
}

__global__ void k_init(int* __restrict__ hist, float* __restrict__ diffs_acc) {
  int i = blockIdx.x * blockDim.x + threadIdx.x;
  if (i < NC_SZ * NE_SZ) hist[i] = 0;
  if (i == 0) *diffs_acc = 0.f;
}

// ---------------------- bf16 WMMA GEMM with double-buffered async-LDS staging
// C[M,N] = op(A[M,K] @ W[N,K]^T + bias), bias read only for col < bias_n.
// Block: 256 threads = 8 waves; block tile 256M x 64N; wave tile 32M x 64N.
// Per 32-K step the block DMAs A(256x32)+B(64x32) bf16 into LDS with
// GLOBAL_LOAD_ASYNC_TO_LDS_B128 (5 async instr / wave / step), double buffered.
template <bool RELU, bool OUT_BF16>
__global__ __launch_bounds__(256)
void k_gemm_wmma(const bf16_t* __restrict__ A, const bf16_t* __restrict__ W,
                 const float* __restrict__ bias, int bias_n,
                 void* __restrict__ out, int M, int N, int K) {
  __shared__ __align__(16) bf16_t smem[2][(256 + 64) * 32];   // 40 KB
  const int tid = threadIdx.x;
  const int lane = tid & 31;
  const int wave = tid >> 5;
  const int m_blk = blockIdx.x * 256;
  const int n_blk = blockIdx.y * 64;

  // async copy coordinates: each thread owns 4 A-chunks + 1 B-chunk of 16B
  const int sub = tid & 3;          // 16B chunk within a 64B (32-elem) row
  const int crow = tid >> 2;        // base row (A: +64*i, B: exact)
  unsigned int sbase[2];
  sbase[0] = (unsigned int)(uintptr_t)&smem[0][0];
  sbase[1] = (unsigned int)(uintptr_t)&smem[1][0];

  const int nk = K >> 5;
  auto issue = [&](int buf, int kb) {
#pragma unroll
    for (int i = 0; i < 4; ++i) {
      int row = crow + 64 * i;
      async_ld16(sbase[buf] + (unsigned int)(row * 64 + sub * 16),
                 A + (size_t)(m_blk + row) * K + kb + sub * 8);
    }
    async_ld16(sbase[buf] + (unsigned int)(256 * 64 + crow * 64 + sub * 16),
               W + (size_t)(n_blk + crow) * K + kb + sub * 8);
  };

  const int m_loc = wave * 32;
  const int akh = (lane >> 4) * 16;  // A K sub-offset per half-wave
  const int fr  = lane & 15;         // fragment row/col within 16
  const int bkh = (lane >> 4) * 16;  // B K sub-offset per half-wave

  v8f acc[2][4] = {};
  issue(0, 0);
  for (int kbi = 0; kbi < nk; ++kbi) {
    int buf = kbi & 1;
    if (kbi + 1 < nk) {
      issue(buf ^ 1, (kbi + 1) << 5);
      asm volatile("s_wait_asynccnt 0x5" ::: "memory");  // oldest 5 (this buf) done
    } else {
      asm volatile("s_wait_asynccnt 0x0" ::: "memory");
    }
    __syncthreads();
    const bf16_t* sA = &smem[buf][0];
    const bf16_t* sB = &smem[buf][256 * 32];
    v16bf bfr[4];
#pragma unroll
    for (int f = 0; f < 4; ++f) {
      const bf16_t* br = sB + (f * 16 + fr) * 32 + bkh;
      bfr[f] = frag16(br, br + 8);
    }
#pragma unroll
    for (int t = 0; t < 2; ++t) {
      const bf16_t* ar = sA + (m_loc + t * 16 + fr) * 32 + akh;
      v16bf af = frag16(ar, ar + 8);
#pragma unroll
      for (int f = 0; f < 4; ++f)
        acc[t][f] = __builtin_amdgcn_wmma_f32_16x16x32_bf16(
            false, af, false, bfr[f], (short)0, acc[t][f], false, false);
    }
    __syncthreads();
  }

  // epilogue: C layout -> row = v + 8*(lane>>4), col = lane&15
  int cin = lane & 15;
#pragma unroll
  for (int t = 0; t < 2; ++t) {
    int rbase = m_blk + m_loc + t * 16 + 8 * (lane >> 4);
#pragma unroll
    for (int f = 0; f < 4; ++f) {
      int col = n_blk + f * 16 + cin;
      float bv = (col < bias_n) ? bias[col] : 0.f;
#pragma unroll
      for (int v = 0; v < 8; ++v) {
        float val = acc[t][f][v] + bv;
        if (RELU) val = fmaxf(val, 0.f);
        size_t idx = (size_t)(rbase + v) * N + col;
        if (OUT_BF16) ((bf16_t*)out)[idx] = (bf16_t)val;
        else          ((float*)out)[idx]  = val;
      }
    }
  }
}

// -------------------- fused codebook distance + argmin + hist + commit-MSE
// grid = (B/16, NC), block = 512 threads (16 waves). Wave w: cols [w*32, w*32+32).
__global__ __launch_bounds__(512)
void k_dist_argmin(const bf16_t* __restrict__ z, const bf16_t* __restrict__ eT_bf,
                   const float* __restrict__ eT_f, const float* __restrict__ e_sq,
                   int* __restrict__ ind, int* __restrict__ hist,
                   float* __restrict__ diffs_acc) {
  __shared__ float s_sc[16][NE_SZ];     // 32 KB
  int c = blockIdx.y;
  int m0 = blockIdx.x * 16;
  int lane = threadIdx.x & 31;
  int wave = threadIdx.x >> 5;

  // --- WMMA phase: scores = z . e_n over K=512
  const bf16_t* Zrow = z + (size_t)(m0 + (lane & 15)) * ED_SZ + c * D_SZ;
  const bf16_t* Eb = eT_bf + (size_t)c * NE_SZ * D_SZ;
  int akh = (lane >> 4) * 16;
  int brow = (lane & 15);
  int bkh = (lane >> 4) * 16;
  v8f acc[2] = {};
  for (int kb = 0; kb < D_SZ; kb += 32) {
    v16bf af = frag16(Zrow + kb + akh, Zrow + kb + akh + 8);
#pragma unroll
    for (int f = 0; f < 2; ++f) {
      const bf16_t* Er = Eb + (size_t)(wave * 32 + f * 16 + brow) * D_SZ + kb + bkh;
      v16bf bfr = frag16(Er, Er + 8);
      acc[f] = __builtin_amdgcn_wmma_f32_16x16x32_bf16(
          false, af, false, bfr, (short)0, acc[f], false, false);
    }
  }
  const float* esq_c = e_sq + c * NE_SZ;
  int rloc = 8 * (lane >> 4);
  int cin = lane & 15;
#pragma unroll
  for (int f = 0; f < 2; ++f) {
#pragma unroll
    for (int v = 0; v < 8; ++v) {
      int col = wave * 32 + f * 16 + cin;
      s_sc[rloc + v][col] = esq_c[col] - 2.0f * acc[f][v];   // ||z||^2 const per row
    }
  }
  __syncthreads();

  // --- argmin: wave r handles row r, first-index tie-break
  float bv = 3.4e38f; int bi = 0;
  for (int j = lane; j < NE_SZ; j += 32) {
    float sv = s_sc[wave][j];
    if (sv < bv) { bv = sv; bi = j; }
  }
#pragma unroll
  for (int off = 16; off > 0; off >>= 1) {
    float ov = __shfl_down(bv, off);
    int   oi = __shfl_down(bi, off);
    if (ov < bv || (ov == bv && oi < bi)) { bv = ov; bi = oi; }
  }
  bi = __shfl(bi, 0);
  int grow = m0 + wave;
  if (lane == 0) {
    ind[(size_t)grow * NC_SZ + c] = bi;
    atomicAdd(&hist[c * NE_SZ + bi], 1);
  }

  // --- commitment MSE contribution: sum_d (e[bi][d] - z[row][d])^2
  const float*  Q  = eT_f + (size_t)c * NE_SZ * D_SZ + (size_t)bi * D_SZ;
  const bf16_t* Zr = z + (size_t)grow * ED_SZ + c * D_SZ;
  float ds = 0.f;
  for (int d = lane; d < D_SZ; d += 32) {
    float dv = Q[d] - (float)Zr[d];
    ds += dv * dv;
  }
#pragma unroll
  for (int off = 16; off > 0; off >>= 1) ds += __shfl_down(ds, off);
  if (lane == 0) atomicAdd(diffs_acc, ds);
}

// ---------------------------------------------------- tiny decoder MLP (fp32)
__global__ __launch_bounds__(256)
void k_decoder(const float* __restrict__ embx, const float* __restrict__ Wd1,
               const float* __restrict__ bd1, const float* __restrict__ Wd2,
               const float* __restrict__ bd2, float* __restrict__ out) {
  __shared__ float sW1[63 * 63];
  __shared__ float sb1[63], sW2[63];
  for (int i = threadIdx.x; i < 63 * 63; i += 256) sW1[i] = Wd1[i];
  if (threadIdx.x < 63) { sb1[threadIdx.x] = bd1[threadIdx.x]; sW2[threadIdx.x] = Wd2[threadIdx.x]; }
  __syncthreads();
  int b = blockIdx.x * 256 + threadIdx.x;
  const float* e = embx + (size_t)b * 64;
  float ev[63];
#pragma unroll
  for (int i = 0; i < 63; ++i) ev[i] = e[i];
  float acc = bd2[0];
#pragma unroll 1
  for (int j = 0; j < 63; ++j) {
    float h = sb1[j];
#pragma unroll
    for (int i = 0; i < 63; ++i) h = fmaf(ev[i], sW1[j * 63 + i], h);
    acc += fmaxf(h, 0.f) * sW2[j];
  }
  out[b] = acc;
}

// --------------------------------------------------- diffs + perplexity
__global__ __launch_bounds__(128)
void k_finalize(const int* __restrict__ hist, const float* __restrict__ diffs_acc,
                float* __restrict__ out) {
  __shared__ float red[128];
  float s = 0.f;
  for (int i = threadIdx.x; i < NC_SZ * NE_SZ; i += 128) {
    float pv = (float)hist[i] * (1.0f / (float)B_SZ);
    s += pv * logf(pv + 1e-10f);
  }
  red[threadIdx.x] = s;
  __syncthreads();
  for (int off = 64; off > 0; off >>= 1) {
    if (threadIdx.x < off) red[threadIdx.x] += red[threadIdx.x + off];
    __syncthreads();
  }
  if (threadIdx.x == 0) {
    out[B_SZ]     = diffs_acc[0] * (1.0f / ((float)B_SZ * (float)D_SZ));
    out[B_SZ + 1] = expf(-red[0]);
  }
}

// ---------------------------------------------------------------- launcher
extern "C" void kernel_launch(void* const* d_in, const int* in_sizes, int n_in,
                              void* d_out, int out_size, void* d_ws, size_t ws_size,
                              hipStream_t stream) {
  const float* x    = (const float*)d_in[0];
  const float* W_f  = (const float*)d_in[1];
  const float* b_f  = (const float*)d_in[2];
  const float* W_x1 = (const float*)d_in[3];
  const float* b_x1 = (const float*)d_in[4];
  const float* W_x2 = (const float*)d_in[5];
  const float* b_x2 = (const float*)d_in[6];
  const float* Wq   = (const float*)d_in[7];
  const float* bq   = (const float*)d_in[8];
  const float* Wd1  = (const float*)d_in[9];
  const float* bd1  = (const float*)d_in[10];
  const float* Wd2  = (const float*)d_in[11];
  const float* bd2  = (const float*)d_in[12];
  const float* embed= (const float*)d_in[13];
  float* out = (float*)d_out;

  char* p = (char*)d_ws;
  size_t off = 0;
  auto take = [&](size_t bytes) -> void* {
    void* r = p + off;
    off = (off + bytes + 255) & ~(size_t)255;
    return r;
  };
  bf16_t* x_bf    = (bf16_t*)take((size_t)B_SZ * IN_SZ * 2);
  bf16_t* Wf_bf   = (bf16_t*)take((size_t)CH_SZ * IN_SZ * 2);
  bf16_t* Wx1_bf  = (bf16_t*)take((size_t)CH_SZ * IN_SZ * 2);
  bf16_t* Wq_bf   = (bf16_t*)take((size_t)ED_SZ * CH_SZ * 2);
  bf16_t* Wx2_bf  = (bf16_t*)take((size_t)64 * CH_SZ * 2);
  bf16_t* eT_bf   = (bf16_t*)take((size_t)NC_SZ * NE_SZ * D_SZ * 2);
  float*  eT_f    = (float*) take((size_t)NC_SZ * NE_SZ * D_SZ * 4);
  float*  e_sq    = (float*) take((size_t)NC_SZ * NE_SZ * 4);
  bf16_t* pre_bf  = (bf16_t*)take((size_t)B_SZ * CH_SZ * 2);
  bf16_t* prex_bf = (bf16_t*)take((size_t)B_SZ * CH_SZ * 2);
  bf16_t* z_bf    = (bf16_t*)take((size_t)B_SZ * ED_SZ * 2);
  float*  embx    = (float*) take((size_t)B_SZ * 64 * 4);
  int*    ind     = (int*)   take((size_t)B_SZ * NC_SZ * 4);
  int*    hist    = (int*)   take((size_t)NC_SZ * NE_SZ * 4);
  float*  dacc    = (float*) take(4);
  (void)ws_size; (void)in_sizes; (void)n_in; (void)out_size;

  // init accumulators (every call; deterministic)
  k_init<<<16, 256, 0, stream>>>(hist, dacc);

  // precision conversion / layout prep
  k_f32_to_bf16<<<4096, 256, 0, stream>>>(x,    x_bf,   B_SZ * IN_SZ);
  k_f32_to_bf16<<<2048, 256, 0, stream>>>(W_f,  Wf_bf,  CH_SZ * IN_SZ);
  k_f32_to_bf16<<<2048, 256, 0, stream>>>(W_x1, Wx1_bf, CH_SZ * IN_SZ);
  k_f32_to_bf16<<<4096, 256, 0, stream>>>(Wq,   Wq_bf,  ED_SZ * CH_SZ);
  k_pad_wx2<<<(64 * CH_SZ + 255) / 256, 256, 0, stream>>>(W_x2, Wx2_bf);
  k_embed_T<<<(NC_SZ * D_SZ * NE_SZ + 255) / 256, 256, 0, stream>>>(embed, eT_bf, eT_f);
  k_esq<<<(NC_SZ * NE_SZ + 255) / 256, 256, 0, stream>>>(eT_f, e_sq);

  // pre  = relu(x @ W_f^T + b_f)      [B, CH]   (bf16 out)
  k_gemm_wmma<true, true><<<dim3(B_SZ / 256, CH_SZ / 64), 256, 0, stream>>>(
      x_bf, Wf_bf, b_f, CH_SZ, pre_bf, B_SZ, CH_SZ, IN_SZ);
  // prex = relu(x @ W_x1^T + b_x1)    [B, CH]   (bf16 out)
  k_gemm_wmma<true, true><<<dim3(B_SZ / 256, CH_SZ / 64), 256, 0, stream>>>(
      x_bf, Wx1_bf, b_x1, CH_SZ, prex_bf, B_SZ, CH_SZ, IN_SZ);
  // z    = pre @ Wq^T + bq            [B, ED]   (bf16 out)
  k_gemm_wmma<false, true><<<dim3(B_SZ / 256, ED_SZ / 64), 256, 0, stream>>>(
      pre_bf, Wq_bf, bq, ED_SZ, z_bf, B_SZ, ED_SZ, CH_SZ);
  // embx = prex @ W_x2p^T + b_x2      [B, 64]   (f32 out, col 63 padded)
  k_gemm_wmma<false, false><<<dim3(B_SZ / 256, 1), 256, 0, stream>>>(
      prex_bf, Wx2_bf, b_x2, 63, embx, B_SZ, 64, CH_SZ);

  // distances + argmin + histogram + commitment MSE
  k_dist_argmin<<<dim3(B_SZ / 16, NC_SZ), 512, 0, stream>>>(
      z_bf, eT_bf, eT_f, e_sq, ind, hist, dacc);

  // decoder -> d_out[0..B)
  k_decoder<<<B_SZ / 256, 256, 0, stream>>>(embx, Wd1, bd1, Wd2, bd2, out);

  // diffs + perplexity -> d_out[B], d_out[B+1]
  k_finalize<<<1, 128, 0, stream>>>(hist, dacc, out);
}
